// TestModel_84662395338815
// MI455X (gfx1250) — compile-verified, matrix operand order checked by reading, not run on hardware
//
#include <hip/hip_runtime.h>

// ---------------------------------------------------------------------------
// Recursive binary-tree shape encoder for MI455X (gfx1250, wave32, WMMA).
//   HIDDEN=512, LATENT=256, SYM=64, BOX=128, LEAF=16384, NODE=32767
// Tree is a perfect heap: node LEAF+j combines nodes 2j and 2j+1 -> 14 levels.
// All GEMMs run on v_wmma_f32_16x16x32_bf16 with 16-node row tiles.
// Activations use the CDNA5 hardware tanh (V_TANH_F32) when available.
// ---------------------------------------------------------------------------

typedef __bf16 bf16;
typedef __attribute__((ext_vector_type(16))) __bf16 v16bf;
typedef __attribute__((ext_vector_type(8)))  float  v8f;

#define LEAFN   16384
#define NODEN   32767
#define LATENT  256
#define HIDDEN  512
#define SYMN    64
#define BOXN    128
#define PAD     8     // bf16 elements of row padding (16B) to break LDS bank conflicts

// ---- fast tanh: single V_TANH_F32 on gfx1250 ------------------------------
__device__ __forceinline__ float fast_tanh(float x) {
#if __has_builtin(__builtin_amdgcn_tanhf)
    return __builtin_amdgcn_tanhf(x);
#elif __has_builtin(__builtin_amdgcn_tanh_f32)
    return __builtin_amdgcn_tanh_f32(x);
#else
    // branchless rational fallback (avoid libm's exec-mask branches)
    float x2 = x * x;
    float p = __builtin_fmaf(x2, __builtin_fmaf(x2, 0.00489352455891786f,
                                                0.0535161226445565f), 0.398942280401432f);
    p = x * __builtin_fmaf(x2, p, 2.38073657977324f);
    float q = __builtin_fmaf(x2, __builtin_fmaf(x2, 0.118098824949826f,
                                                0.898934843013341f), 2.38073657977324f);
    float r = p / q;
    return fminf(fmaxf(r, -1.0f), 1.0f);
#endif
}

__device__ __forceinline__ v8f wmma_bf16(v16bf a, v16bf b, v8f c) {
    return __builtin_amdgcn_wmma_f32_16x16x32_bf16(
        /*neg_a=*/false, a, /*neg_b=*/false, b,
        /*c_mod=*/(short)0, c, /*reuse_a=*/false, /*reuse_b=*/false);
}

// Load one lane's 16-element fragment for a 16x32 (bf16) WMMA operand.
// `row` points to the start of this lane's row/col at K-base; layout per ISA:
//   lanes 0-15  hold K = {0..7, 16..23} of the 32-wide slice
//   lanes 16-31 hold K = {8..15, 24..31}
__device__ __forceinline__ v16bf load_frag(const bf16* row, int halfOff) {
    v16bf v;
#pragma unroll
    for (int e = 0; e < 8; ++e) v[e] = row[halfOff + e];
#pragma unroll
    for (int e = 0; e < 8; ++e) v[8 + e] = row[16 + halfOff + e];
    return v;
}

// ---------------------------------------------------------------------------
// Weight prep: fp32 [rows][cols] -> bf16 transposed [cols][rows] (N-major, K
// contiguous) so WMMA B-fragments are two contiguous 16B lane reads.
// ---------------------------------------------------------------------------
__global__ void convert_T(const float* __restrict__ src, bf16* __restrict__ dst,
                          int rows, int cols) {
    int idx = blockIdx.x * blockDim.x + threadIdx.x;
    if (idx < rows * cols) {
        int r = idx / cols, c = idx % cols;
        dst[(size_t)c * rows + r] = (bf16)src[idx];
    }
}

// ---------------------------------------------------------------------------
// Leaf encoder: h[leaf] = tanh(shape[:,leaf] @ Wb + bb)   (16 leaves / block)
// shape is [BOX][LEAF] (leaf contiguous) -> transpose-stage into LDS.
// ---------------------------------------------------------------------------
__global__ __launch_bounds__(256) void leaf_kernel(
    const float* __restrict__ shape, const bf16* __restrict__ WbT,
    const float* __restrict__ bb, bf16* __restrict__ hBuf) {
    __shared__ bf16 As[16][BOXN + PAD];
    const int t = threadIdx.x;
    const int leaf0 = blockIdx.x * 16;

    // stage: thread t covers leaf m = t%16, k-range [ (t/16)*8 , +8 )
    {
        const int m = t % 16, kidx = t / 16;
#pragma unroll
        for (int q = 0; q < 8; ++q) {
            int k = kidx * 8 + q;
            As[m][k] = (bf16)shape[(size_t)k * LEAFN + leaf0 + m];
        }
    }
    __syncthreads();

    const int wave = t / 32, lane = t % 32;
    const int mrow = lane % 16;
    const int halfOff = (lane >= 16) ? 8 : 0;
    const int mAdd = (lane >= 16) ? 8 : 0;

    // N=256 -> 16 n-tiles; 8 waves x 2 tiles
#pragma unroll
    for (int u = 0; u < 2; ++u) {
        const int nt = wave * 2 + u;
        const int col = nt * 16 + mrow;
        const float bias = bb[col];
        v8f acc;
#pragma unroll
        for (int r = 0; r < 8; ++r) acc[r] = bias;
#pragma unroll
        for (int kb = 0; kb < BOXN / 32; ++kb) {
            v16bf aF = load_frag(&As[mrow][kb * 32], halfOff);
            v16bf bF = load_frag(&WbT[(size_t)col * BOXN + kb * 32], halfOff);
            acc = wmma_bf16(aF, bF, acc);
        }
#pragma unroll
        for (int r = 0; r < 8; ++r) {
            int leaf = leaf0 + r + mAdd;
            hBuf[(size_t)leaf * LATENT + col] = (bf16)fast_tanh(acc[r]);
        }
    }
}

// ---------------------------------------------------------------------------
// One tree level: 16 internal nodes per block, both MLP paths via WMMA,
// per-row select by k2 flag.
// ---------------------------------------------------------------------------
__global__ __launch_bounds__(256) void level_kernel(
    const int* __restrict__ kids, const float* __restrict__ param,
    bf16* __restrict__ hBuf,
    const bf16* __restrict__ WlT, const bf16* __restrict__ WrT,
    const float* __restrict__ b1,
    const bf16* __restrict__ W2T, const float* __restrict__ b2,
    const bf16* __restrict__ WslT, const bf16* __restrict__ WsrT,
    const float* __restrict__ bs1,
    const bf16* __restrict__ Ws2T, const float* __restrict__ bs2,
    int jBase, int jEnd, float* __restrict__ out_root) {
    __shared__ bf16 As[16][LATENT + PAD];   // left child latent
    __shared__ bf16 Bs[16][LATENT + PAD];   // right child latent
    __shared__ bf16 Ss[16][SYMN + PAD];     // sym params of left child
    __shared__ bf16 Mid1[16][HIDDEN + PAD]; // std-path hidden
    __shared__ bf16 Mid2[16][HIDDEN + PAD]; // sym-path hidden
    __shared__ int  flag[16];

    const int t = threadIdx.x;
    const int j0 = jBase + blockIdx.x * 16;

    // ---- stage children: thread t = (row m = t/16, chunk c = t%16 of 16 elems)
    {
        const int m = t / 16, c = t % 16;
        const int j = j0 + m;
        if (j < jEnd) {
            const int i = LEAFN + j;
            const int k0 = kids[3 * i + 0];
            const int k1 = kids[3 * i + 1];
            const int k2 = kids[3 * i + 2];
            if (c == 0) flag[m] = k2;
            const bf16* aRow = hBuf + (size_t)(k0 - 1) * LATENT;
            const bf16* bRow = hBuf + (size_t)(k1 - 1) * LATENT;
#pragma unroll
            for (int q = 0; q < 16; ++q) {
                As[m][c * 16 + q] = aRow[c * 16 + q];
                Bs[m][c * 16 + q] = bRow[c * 16 + q];
            }
            if (c < 4) {
                const float* sRow = param + (size_t)(k0 - 1) * SYMN;
#pragma unroll
                for (int q = 0; q < 16; ++q) Ss[m][c * 16 + q] = (bf16)sRow[c * 16 + q];
            }
        } else {
            if (c == 0) flag[m] = 0;
#pragma unroll
            for (int q = 0; q < 16; ++q) {
                As[m][c * 16 + q] = (bf16)0.0f;
                Bs[m][c * 16 + q] = (bf16)0.0f;
            }
            if (c < 4) {
#pragma unroll
                for (int q = 0; q < 16; ++q) Ss[m][c * 16 + q] = (bf16)0.0f;
            }
        }
    }
    __syncthreads();

    const int wave = t / 32, lane = t % 32;
    const int mrow = lane % 16;
    const int halfOff = (lane >= 16) ? 8 : 0;
    const int mAdd = (lane >= 16) ? 8 : 0;

    // ---- layer 1: 64 units = (2 paths) x (32 n-tiles over HIDDEN); 8/wave
#pragma unroll
    for (int q = 0; q < 8; ++q) {
        const int u = wave * 8 + q;
        const int path = u >> 5;
        const int nt = u & 31;
        const int col = nt * 16 + mrow;
        if (path == 0) {
            const float bias = b1[col];
            v8f acc;
#pragma unroll
            for (int r = 0; r < 8; ++r) acc[r] = bias;
#pragma unroll
            for (int kb = 0; kb < LATENT / 32; ++kb) {
                v16bf aF = load_frag(&As[mrow][kb * 32], halfOff);
                v16bf wF = load_frag(&WlT[(size_t)col * LATENT + kb * 32], halfOff);
                acc = wmma_bf16(aF, wF, acc);
            }
#pragma unroll
            for (int kb = 0; kb < LATENT / 32; ++kb) {
                v16bf aF = load_frag(&Bs[mrow][kb * 32], halfOff);
                v16bf wF = load_frag(&WrT[(size_t)col * LATENT + kb * 32], halfOff);
                acc = wmma_bf16(aF, wF, acc);
            }
#pragma unroll
            for (int r = 0; r < 8; ++r) Mid1[r + mAdd][col] = (bf16)fast_tanh(acc[r]);
        } else {
            const float bias = bs1[col];
            v8f acc;
#pragma unroll
            for (int r = 0; r < 8; ++r) acc[r] = bias;
#pragma unroll
            for (int kb = 0; kb < LATENT / 32; ++kb) {
                v16bf aF = load_frag(&As[mrow][kb * 32], halfOff);
                v16bf wF = load_frag(&WslT[(size_t)col * LATENT + kb * 32], halfOff);
                acc = wmma_bf16(aF, wF, acc);
            }
#pragma unroll
            for (int kb = 0; kb < SYMN / 32; ++kb) {
                v16bf aF = load_frag(&Ss[mrow][kb * 32], halfOff);
                v16bf wF = load_frag(&WsrT[(size_t)col * SYMN + kb * 32], halfOff);
                acc = wmma_bf16(aF, wF, acc);
            }
#pragma unroll
            for (int r = 0; r < 8; ++r) Mid2[r + mAdd][col] = (bf16)fast_tanh(acc[r]);
        }
    }
    __syncthreads();

    // ---- layer 2: 16 n-tiles over LATENT; both paths per unit, select by flag
#pragma unroll
    for (int q = 0; q < 2; ++q) {
        const int nt = wave * 2 + q;
        const int col = nt * 16 + mrow;
        const float bias1 = b2[col], bias2 = bs2[col];
        v8f acc1, acc2;
#pragma unroll
        for (int r = 0; r < 8; ++r) { acc1[r] = bias1; acc2[r] = bias2; }
#pragma unroll
        for (int kb = 0; kb < HIDDEN / 32; ++kb) {
            v16bf m1 = load_frag(&Mid1[mrow][kb * 32], halfOff);
            v16bf w1 = load_frag(&W2T[(size_t)col * HIDDEN + kb * 32], halfOff);
            acc1 = wmma_bf16(m1, w1, acc1);
            v16bf m2 = load_frag(&Mid2[mrow][kb * 32], halfOff);
            v16bf w2 = load_frag(&Ws2T[(size_t)col * HIDDEN + kb * 32], halfOff);
            acc2 = wmma_bf16(m2, w2, acc2);
        }
#pragma unroll
        for (int r = 0; r < 8; ++r) {
            const int m = r + mAdd;
            const int j = j0 + m;
            if (j < jEnd) {
                const int i = LEAFN + j;
                const float v1 = fast_tanh(acc1[r]);
                const float v2 = fast_tanh(acc2[r]);
                const float v = flag[m] ? v2 : v1;
                hBuf[(size_t)i * LATENT + col] = (bf16)v;
                if (i == NODEN - 1) out_root[col] = v;   // fp32 root latent
            }
        }
    }
}

// ---------------------------------------------------------------------------
// Host-side orchestration (graph-capture safe: only async launches on stream)
// ---------------------------------------------------------------------------
extern "C" void kernel_launch(void* const* d_in, const int* in_sizes, int n_in,
                              void* d_out, int out_size, void* d_ws, size_t ws_size,
                              hipStream_t stream) {
    const int*   kids  = (const int*)  d_in[0];
    const float* shape = (const float*)d_in[1];
    const float* param = (const float*)d_in[2];
    const float* Wb    = (const float*)d_in[3];
    const float* bb    = (const float*)d_in[4];
    const float* Wl    = (const float*)d_in[5];
    const float* Wr    = (const float*)d_in[6];
    const float* b1    = (const float*)d_in[7];
    const float* W2    = (const float*)d_in[8];
    const float* b2    = (const float*)d_in[9];
    const float* Wsl   = (const float*)d_in[10];
    const float* Wsr   = (const float*)d_in[11];
    const float* bs1   = (const float*)d_in[12];
    const float* Ws2   = (const float*)d_in[13];
    const float* bs2   = (const float*)d_in[14];
    float* out = (float*)d_out;

    // workspace carve-up
    char* ws = (char*)d_ws;
    bf16* hBuf = (bf16*)ws;            ws += (size_t)32768 * LATENT * sizeof(bf16);
    bf16* WbT  = (bf16*)ws;            ws += (size_t)BOXN   * LATENT * sizeof(bf16);
    bf16* WlT  = (bf16*)ws;            ws += (size_t)LATENT * HIDDEN * sizeof(bf16);
    bf16* WrT  = (bf16*)ws;            ws += (size_t)LATENT * HIDDEN * sizeof(bf16);
    bf16* W2T  = (bf16*)ws;            ws += (size_t)HIDDEN * LATENT * sizeof(bf16);
    bf16* WslT = (bf16*)ws;            ws += (size_t)LATENT * HIDDEN * sizeof(bf16);
    bf16* WsrT = (bf16*)ws;            ws += (size_t)SYMN   * HIDDEN * sizeof(bf16);
    bf16* Ws2T = (bf16*)ws;            ws += (size_t)HIDDEN * LATENT * sizeof(bf16);

    // 1) weights -> bf16, transposed to [N][K]
    auto cvt = [&](const float* src, bf16* dst, int r, int c) {
        int n = r * c;
        convert_T<<<(n + 255) / 256, 256, 0, stream>>>(src, dst, r, c);
    };
    cvt(Wb,  WbT,  BOXN,   LATENT);
    cvt(Wl,  WlT,  LATENT, HIDDEN);
    cvt(Wr,  WrT,  LATENT, HIDDEN);
    cvt(W2,  W2T,  HIDDEN, LATENT);
    cvt(Wsl, WslT, LATENT, HIDDEN);
    cvt(Wsr, WsrT, SYMN,   HIDDEN);
    cvt(Ws2, Ws2T, HIDDEN, LATENT);

    // 2) leaf encoder: 16384 leaves, 16 per block
    leaf_kernel<<<LEAFN / 16, 256, 0, stream>>>(shape, WbT, bb, hBuf);

    // 3) 14 tree levels (8192, 4096, ..., 1 nodes); heap layout => contiguous j
    int jBase = 0;
    for (int cnt = LEAFN / 2; cnt >= 1; cnt >>= 1) {
        const int jEnd = jBase + cnt;
        const int blocks = (cnt + 15) / 16;
        level_kernel<<<blocks, 256, 0, stream>>>(
            kids, param, hBuf,
            WlT, WrT, b1, W2T, b2, WslT, WsrT, bs1, Ws2T, bs2,
            jBase, jEnd, out);
        jBase = jEnd;
    }
}
